// PoseEncoder_63960652972189
// MI455X (gfx1250) — matrix operand order repaired and checked by reading, not compiled, not run on hardware
//
#include <hip/hip_runtime.h>
#include <hip/hip_bf16.h>

typedef __attribute__((ext_vector_type(16))) _Float16 v16h;
typedef __attribute__((ext_vector_type(8)))  _Float16 v8h;
typedef __attribute__((ext_vector_type(8)))  float    v8f;

#define BATCH 16
#define NPTS  8192

// ---------------------------------------------------------------------------
// Generic fused pointwise-MLP layer:  Y = relu((A @ W^T + b) * s + o)
// A: (M, Kp) f16 row-major (Kp % 32 == 0, zero padded)
// Wh: (Cout, Kp) f16 row-major (zero padded)
// mode 0: f16 out, stride Yld.  mode 1: head2 -> f32, transposed (B,64,N) out.
// Each wave owns a 16(M) x 16*NT(Cout) tile; K swept in steps of 32 with
// v_wmma_f32_16x16x32_f16. VGPR layouts per CDNA5 ISA 7.12.2:
//  A frag lane L (g=L/16, m=L%16): halves 0..7 -> K=k0+g*8+h, 8..15 -> +16
//  B frag lane L: n = L%16, same K layout as A frag
//  C/D lane L vgpr r: n = L%16, m = (L/16)*8 + r
// ---------------------------------------------------------------------------
template <int NT>
__global__ __launch_bounds__(256)
void mlp_layer_kernel(const _Float16* __restrict__ A, const _Float16* __restrict__ Wh,
                      const float* __restrict__ bias, const float* __restrict__ scale,
                      const float* __restrict__ offs,
                      _Float16* __restrict__ Y, float* __restrict__ Yf,
                      int M, int Kp, int Yld, int nGroups, int mode)
{
    int lane = threadIdx.x & 31;
    int wv   = (blockIdx.x * blockDim.x + threadIdx.x) >> 5;
    int totalWaves = (M >> 4) * nGroups;
    if (wv >= totalWaves) return;                    // wave-uniform exit
    int mtile = (wv / nGroups) << 4;
    int n0    = (wv % nGroups) * (16 * NT);
    int g  = lane >> 4;
    int mr = lane & 15;

    v8f acc[NT];
#pragma unroll
    for (int j = 0; j < NT; ++j)
#pragma unroll
        for (int r = 0; r < 8; ++r) acc[j][r] = 0.0f;

    const _Float16* ap = A + (size_t)(mtile + mr) * Kp + g * 8;
    for (int k0 = 0; k0 < Kp; k0 += 32) {
        union { v16h v; v8h h[2]; } a;
        a.h[0] = *(const v8h*)(ap + k0);
        a.h[1] = *(const v8h*)(ap + k0 + 16);
#pragma unroll
        for (int j = 0; j < NT; ++j) {
            const _Float16* bp = Wh + (size_t)(n0 + j * 16 + mr) * Kp + k0 + g * 8;
            union { v16h v; v8h h[2]; } bu;
            bu.h[0] = *(const v8h*)bp;
            bu.h[1] = *(const v8h*)(bp + 16);
            v16h bfrag = bu.v;
            acc[j] = __builtin_amdgcn_wmma_f32_16x16x32_f16(
                false, a.v, false, bfrag, (short)0, acc[j], false, false);
        }
    }

#pragma unroll
    for (int j = 0; j < NT; ++j) {
        int n = n0 + j * 16 + mr;
        float bb = bias  ? bias[n]  : 0.0f;
        float ss = scale ? scale[n] : 1.0f;
        float oo = offs  ? offs[n]  : 0.0f;
#pragma unroll
        for (int r = 0; r < 8; ++r) {
            int m = mtile + g * 8 + r;
            float v = acc[j][r] + bb;
            if (mode == 0) {
                v = v * ss + oo;
                v = v > 0.0f ? v : 0.0f;
                Y[(size_t)m * Yld + n] = (_Float16)v;
            } else {
                // m = b*NPTS + p ; out[b, n, p]
                Yf[((size_t)(m >> 13) * 64 + n) * NPTS + (m & (NPTS - 1))] = v;
            }
        }
    }
}

// f32 (Cout,Cin) -> f16 (Cout,Kp) zero-padded
__global__ void convw_kernel(const float* __restrict__ src, _Float16* __restrict__ dst,
                             int cin, int kp, int tot)
{
    int tid = blockIdx.x * blockDim.x + threadIdx.x;
    if (tid >= tot) return;
    int n = tid / kp, k = tid % kp;
    dst[tid] = (k < cin) ? (_Float16)src[(size_t)n * cin + k] : (_Float16)0.0f;
}

// Farthest point sampling: one block per batch, dists live in LDS.
__global__ __launch_bounds__(256)
void fps_kernel(const float* __restrict__ xyz, int* __restrict__ out, int Npts, int npoint)
{
    int b = blockIdx.x;
    const float* p = xyz + (size_t)b * Npts * 3;
    int* idx = out + (size_t)b * npoint;
    __shared__ float sdist[NPTS];
    __shared__ float sv[256];
    __shared__ int   si[256];
    int t = threadIdx.x;
    for (int i = t; i < Npts; i += 256) sdist[i] = 1e10f;
    __syncthreads();
    int last = 0;
    for (int it = 0; it < npoint; ++it) {
        if (t == 0) idx[it] = last;
        float lx = p[last * 3 + 0], ly = p[last * 3 + 1], lz = p[last * 3 + 2];
        float best = -1.0f; int bi = 0;
        for (int i = t; i < Npts; i += 256) {
            float dx = p[i * 3] - lx, dy = p[i * 3 + 1] - ly, dz = p[i * 3 + 2] - lz;
            float d  = dx * dx + dy * dy + dz * dz;
            float nd = fminf(sdist[i], d);
            sdist[i] = nd;
            if (nd > best) { best = nd; bi = i; }
        }
        sv[t] = best; si[t] = bi;
        __syncthreads();
        for (int s = 128; s > 0; s >>= 1) {
            if (t < s) {
                if (sv[t + s] > sv[t] || (sv[t + s] == sv[t] && si[t + s] < si[t])) {
                    sv[t] = sv[t + s]; si[t] = si[t + s];
                }
            }
            __syncthreads();
        }
        last = si[0];
        __syncthreads();
    }
}

__global__ void gather_pts_kernel(const float* __restrict__ src, const int* __restrict__ idx,
                                  float* __restrict__ dst, int Nsrc, int np, int total)
{
    int tid = blockIdx.x * blockDim.x + threadIdx.x;
    if (tid >= total) return;
    int b = tid / np;
    int j = idx[tid];
    const float* s = src + ((size_t)b * Nsrc + j) * 3;
    float* d = dst + (size_t)tid * 3;
    d[0] = s[0]; d[1] = s[1]; d[2] = s[2];
}

// Collect first (lowest-index) 16 points within radius; pad with first hit.
__global__ void ball_query_kernel(const float* __restrict__ pts, const float* __restrict__ ctr,
                                  int* __restrict__ out, int Nsrc, int nc, int total, float r2)
{
    int tid = blockIdx.x * blockDim.x + threadIdx.x;
    if (tid >= total) return;
    int b = tid / nc;
    const float* p = pts + (size_t)b * Nsrc * 3;
    float cx = ctr[(size_t)tid * 3], cy = ctr[(size_t)tid * 3 + 1], cz = ctr[(size_t)tid * 3 + 2];
    int tmp[16]; int cnt = 0;
    for (int i = 0; i < Nsrc && cnt < 16; ++i) {
        float dx = p[i * 3] - cx, dy = p[i * 3 + 1] - cy, dz = p[i * 3 + 2] - cz;
        if (dx * dx + dy * dy + dz * dz < r2) tmp[cnt++] = i;
    }
    int f = (cnt > 0) ? tmp[0] : 0;
    for (int k = cnt; k < 16; ++k) tmp[k] = f;
    for (int k = 0; k < 16; ++k) out[(size_t)tid * 16 + k] = tmp[k];
}

// g1 rows: ((b*512+s)*16+j) -> [xyz[idx]-ctr (3), zero pad] width Kp
__global__ void build_g1_kernel(const float* __restrict__ xyz, const float* __restrict__ ctr,
                                const int* __restrict__ nidx, _Float16* __restrict__ out,
                                int Nsrc, int np, int Kp, int total)
{
    int tid = blockIdx.x * blockDim.x + threadIdx.x;
    if (tid >= total) return;
    int s  = (tid >> 4) % np;
    int b  = tid / (np * 16);
    int gi = nidx[tid];
    const float* p = xyz + ((size_t)b * Nsrc + gi) * 3;
    const float* c = ctr + ((size_t)b * np + s) * 3;
    _Float16* o = out + (size_t)tid * Kp;
    o[0] = (_Float16)(p[0] - c[0]);
    o[1] = (_Float16)(p[1] - c[1]);
    o[2] = (_Float16)(p[2] - c[2]);
    for (int k = 3; k < Kp; ++k) o[k] = (_Float16)0.0f;
}

// g2 rows: [l1xyz[idx]-l2xyz (3), l1f[idx] (64), pad->96]
__global__ void build_g2_kernel(const float* __restrict__ l1xyz, const float* __restrict__ l2xyz,
                                const int* __restrict__ nidx, const _Float16* __restrict__ l1f,
                                _Float16* __restrict__ out, int Nsrc, int np, int Cf, int Kp, int total)
{
    int tid = blockIdx.x * blockDim.x + threadIdx.x;
    if (tid >= total) return;
    int s  = (tid >> 4) % np;
    int b  = tid / (np * 16);
    int gi = nidx[tid];
    const float* p = l1xyz + ((size_t)b * Nsrc + gi) * 3;
    const float* c = l2xyz + ((size_t)b * np + s) * 3;
    const _Float16* f = l1f + ((size_t)b * Nsrc + gi) * Cf;
    _Float16* o = out + (size_t)tid * Kp;
    o[0] = (_Float16)(p[0] - c[0]);
    o[1] = (_Float16)(p[1] - c[1]);
    o[2] = (_Float16)(p[2] - c[2]);
    for (int k = 0; k < Cf; ++k) o[3 + k] = f[k];
    for (int k = 3 + Cf; k < Kp; ++k) o[k] = (_Float16)0.0f;
}

// g3 rows (b*128+s): [l2xyz (3), l2f (128), pad->160]
__global__ void build_g3_kernel(const float* __restrict__ l2xyz, const _Float16* __restrict__ l2f,
                                _Float16* __restrict__ out, int Kp, int total)
{
    int tid = blockIdx.x * blockDim.x + threadIdx.x;
    if (tid >= total) return;
    const float* c = l2xyz + (size_t)tid * 3;
    const _Float16* f = l2f + (size_t)tid * 128;
    _Float16* o = out + (size_t)tid * Kp;
    o[0] = (_Float16)c[0]; o[1] = (_Float16)c[1]; o[2] = (_Float16)c[2];
    for (int k = 0; k < 128; ++k) o[3 + k] = f[k];
    for (int k = 131; k < Kp; ++k) o[k] = (_Float16)0.0f;
}

// fp3 input rows (b*128+s): [l3f[b] (256), l2f[row] (128)]
__global__ void build_fp3in_kernel(const _Float16* __restrict__ l3f, const _Float16* __restrict__ l2f,
                                   _Float16* __restrict__ out, int total)
{
    int tid = blockIdx.x * blockDim.x + threadIdx.x;
    if (tid >= total) return;
    int b = tid / 128;
    _Float16* o = out + (size_t)tid * 384;
    const _Float16* a = l3f + (size_t)b * 256;
    const _Float16* f = l2f + (size_t)tid * 128;
    for (int k = 0; k < 256; ++k) o[k] = a[k];
    for (int k = 0; k < 128; ++k) o[256 + k] = f[k];
}

__global__ void maxpool_kernel(const _Float16* __restrict__ X, _Float16* __restrict__ Y,
                               int C, int group, int total)
{
    int tid = blockIdx.x * blockDim.x + threadIdx.x;
    if (tid >= total) return;
    int ro = tid / C, c = tid % C;
    const _Float16* x = X + (size_t)ro * group * C + c;
    float m = (float)x[0];
    for (int j = 1; j < group; ++j) m = fmaxf(m, (float)x[(size_t)j * C]);
    Y[(size_t)ro * C + c] = (_Float16)m;
}

// 3-NN inverse-distance interpolation of 256-ch features.
__global__ void three_interp_kernel(const float* __restrict__ unk, const float* __restrict__ knw,
                                    const _Float16* __restrict__ feats, _Float16* __restrict__ out,
                                    int Nu, int Nk, int C, int outLd, int total)
{
    int tid = blockIdx.x * blockDim.x + threadIdx.x;
    if (tid >= total) return;
    int b = tid / Nu;
    const float* u = unk + (size_t)tid * 3;
    const float* k = knw + (size_t)b * Nk * 3;
    float ux = u[0], uy = u[1], uz = u[2];
    float d0 = 1e30f, d1 = 1e30f, d2 = 1e30f;
    int   i0 = 0, i1 = 0, i2 = 0;
    for (int i = 0; i < Nk; ++i) {
        float dx = k[i * 3] - ux, dy = k[i * 3 + 1] - uy, dz = k[i * 3 + 2] - uz;
        float d = dx * dx + dy * dy + dz * dz;
        if (d < d0)      { d2 = d1; i2 = i1; d1 = d0; i1 = i0; d0 = d; i0 = i; }
        else if (d < d1) { d2 = d1; i2 = i1; d1 = d;  i1 = i; }
        else if (d < d2) { d2 = d;  i2 = i; }
    }
    float w0 = 1.0f / (fmaxf(d0, 0.0f) + 1e-8f);
    float w1 = 1.0f / (fmaxf(d1, 0.0f) + 1e-8f);
    float w2 = 1.0f / (fmaxf(d2, 0.0f) + 1e-8f);
    float ws = w0 + w1 + w2;
    w0 /= ws; w1 /= ws; w2 /= ws;
    const _Float16* f0 = feats + ((size_t)b * Nk + i0) * C;
    const _Float16* f1 = feats + ((size_t)b * Nk + i1) * C;
    const _Float16* f2 = feats + ((size_t)b * Nk + i2) * C;
    _Float16* o = out + (size_t)tid * outLd;
    for (int c = 0; c < C; ++c)
        o[c] = (_Float16)(w0 * (float)f0[c] + w1 * (float)f1[c] + w2 * (float)f2[c]);
}

// append src (rows x C, stride srcLd) into dst at column offset dstOff
__global__ void append_feats_kernel(const _Float16* __restrict__ src, _Float16* __restrict__ dst,
                                    int C, int srcLd, int dstLd, int dstOff, int total)
{
    int tid = blockIdx.x * blockDim.x + threadIdx.x;
    if (tid >= total) return;
    int row = tid / C, c = tid % C;
    dst[(size_t)row * dstLd + dstOff + c] = src[(size_t)row * srcLd + c];
}

// ---------------------------------------------------------------------------
extern "C" void kernel_launch(void* const* d_in, const int* in_sizes, int n_in,
                              void* d_out, int out_size, void* d_ws, size_t ws_size,
                              hipStream_t stream)
{
    (void)n_in; (void)out_size; (void)ws_size;
    auto in = [&](int i) -> const float* { return (const float*)d_in[i]; };

    struct Lay { const float* W; const float* b; const float* s; const float* o; };
    Lay sa1[3], sa2[2], sa3[2], fp3[2], fp2[2], fp1[3], h1, h2;
    const float* xyz;

    bool insertionOrder = (in_sizes[0] == BATCH * NPTS * 3);
    if (insertionOrder) {
        // dict insertion order: xyz, then sa1..head2; layer dicts as W,b,s,o
        auto L = [&](int base) { return Lay{ in(base), in(base + 1), in(base + 2), in(base + 3) }; };
        xyz = in(0);
        sa1[0] = L(1);  sa1[1] = L(5);  sa1[2] = L(9);
        sa2[0] = L(13); sa2[1] = L(17);
        sa3[0] = L(21); sa3[1] = L(25);
        fp3[0] = L(29); fp3[1] = L(33);
        fp2[0] = L(37); fp2[1] = L(41);
        fp1[0] = L(45); fp1[1] = L(49); fp1[2] = L(53);
        h1 = L(57);
        h2 = Lay{ in(61), in(62), nullptr, nullptr };
    } else {
        // jax tree_leaves: sorted keys (params then xyz); layer dicts W,b,o,s
        auto L = [&](int base) { return Lay{ in(base), in(base + 1), in(base + 3), in(base + 2) }; };
        fp1[0] = L(0);  fp1[1] = L(4);  fp1[2] = L(8);
        fp2[0] = L(12); fp2[1] = L(16);
        fp3[0] = L(20); fp3[1] = L(24);
        h1 = L(28);
        h2 = Lay{ in(32), in(33), nullptr, nullptr };
        sa1[0] = L(34); sa1[1] = L(38); sa1[2] = L(42);
        sa2[0] = L(46); sa2[1] = L(50);
        sa3[0] = L(54); sa3[1] = L(58);
        xyz = in(62);
    }

    // ---- workspace bump allocator ----
    size_t off = 0;
    auto alloc = [&](size_t bytes) -> void* {
        off = (off + 255) & ~(size_t)255;
        void* p = (char*)d_ws + off;
        off += bytes;
        return p;
    };
    auto makeW = [&](const Lay& l, int cout, int cin, int kp) -> const _Float16* {
        _Float16* d = (_Float16*)alloc((size_t)cout * kp * sizeof(_Float16));
        int tot = cout * kp;
        convw_kernel<<<(tot + 255) / 256, 256, 0, stream>>>(l.W, d, cin, kp, tot);
        return d;
    };

    // f16 weights (K padded to %32)
    const _Float16* sa1W0 = makeW(sa1[0], 32, 3, 32);
    const _Float16* sa1W1 = makeW(sa1[1], 32, 32, 32);
    const _Float16* sa1W2 = makeW(sa1[2], 64, 32, 32);
    const _Float16* sa2W0 = makeW(sa2[0], 64, 67, 96);
    const _Float16* sa2W1 = makeW(sa2[1], 128, 64, 64);
    const _Float16* sa3W0 = makeW(sa3[0], 128, 131, 160);
    const _Float16* sa3W1 = makeW(sa3[1], 256, 128, 128);
    const _Float16* fp3W0 = makeW(fp3[0], 256, 384, 384);
    const _Float16* fp3W1 = makeW(fp3[1], 256, 256, 256);
    const _Float16* fp2W0 = makeW(fp2[0], 256, 320, 320);
    const _Float16* fp2W1 = makeW(fp2[1], 256, 256, 256);
    const _Float16* fp1W0 = makeW(fp1[0], 128, 256, 256);
    const _Float16* fp1W1 = makeW(fp1[1], 128, 128, 128);
    const _Float16* fp1W2 = makeW(fp1[2], 128, 128, 128);
    const _Float16* h1W   = makeW(h1, 128, 128, 128);
    const _Float16* h2W   = makeW(h2, 64, 128, 128);

    // activation / index buffers
    int*       fidx1  = (int*)alloc(BATCH * 512 * 4);
    float*     l1xyz  = (float*)alloc(BATCH * 512 * 3 * 4);
    int*       nidx1  = (int*)alloc((size_t)BATCH * 512 * 16 * 4);
    _Float16*  g1A    = (_Float16*)alloc((size_t)131072 * 32 * 2);
    _Float16*  t1a    = (_Float16*)alloc((size_t)131072 * 32 * 2);
    _Float16*  t1c    = (_Float16*)alloc((size_t)131072 * 64 * 2);
    _Float16*  l1f    = (_Float16*)alloc((size_t)8192 * 64 * 2);
    int*       fidx2  = (int*)alloc(BATCH * 128 * 4);
    float*     l2xyz  = (float*)alloc(BATCH * 128 * 3 * 4);
    int*       nidx2  = (int*)alloc((size_t)BATCH * 128 * 16 * 4);
    _Float16*  g2A    = (_Float16*)alloc((size_t)32768 * 96 * 2);
    _Float16*  t2a    = (_Float16*)alloc((size_t)32768 * 64 * 2);
    _Float16*  t2b    = (_Float16*)alloc((size_t)32768 * 128 * 2);
    _Float16*  l2f    = (_Float16*)alloc((size_t)2048 * 128 * 2);
    _Float16*  g3A    = (_Float16*)alloc((size_t)2048 * 160 * 2);
    _Float16*  t3a    = (_Float16*)alloc((size_t)2048 * 128 * 2);
    _Float16*  t3b    = (_Float16*)alloc((size_t)2048 * 256 * 2);
    _Float16*  l3f    = (_Float16*)alloc((size_t)16 * 256 * 2);
    _Float16*  fp3in  = (_Float16*)alloc((size_t)2048 * 384 * 2);
    _Float16*  fp3t   = (_Float16*)alloc((size_t)2048 * 256 * 2);
    _Float16*  f2buf  = (_Float16*)alloc((size_t)2048 * 256 * 2);
    _Float16*  fp2in  = (_Float16*)alloc((size_t)8192 * 320 * 2);
    _Float16*  fp2t   = (_Float16*)alloc((size_t)8192 * 256 * 2);
    _Float16*  f1buf  = (_Float16*)alloc((size_t)8192 * 256 * 2);
    _Float16*  fp1in  = (_Float16*)alloc((size_t)131072 * 256 * 2);
    _Float16*  bigA   = (_Float16*)alloc((size_t)131072 * 128 * 2);
    _Float16*  bigB   = (_Float16*)alloc((size_t)131072 * 128 * 2);

    auto layer = [&](const _Float16* A, const _Float16* Wh, const Lay& l,
                     _Float16* Y, float* Yf, int M, int Kp, int Cout, int Yld, int mode) {
        int NT = (Cout >= 64) ? 4 : 2;         // Cout in {32,64,128,256}
        int nGroups = Cout / (16 * NT);
        long waves = (long)(M / 16) * nGroups;
        int blocks = (int)((waves * 32 + 255) / 256);
        if (NT == 4)
            mlp_layer_kernel<4><<<blocks, 256, 0, stream>>>(A, Wh, l.b, l.s, l.o, Y, Yf, M, Kp, Yld, nGroups, mode);
        else
            mlp_layer_kernel<2><<<blocks, 256, 0, stream>>>(A, Wh, l.b, l.s, l.o, Y, Yf, M, Kp, Yld, nGroups, mode);
    };
    auto g1d = [](int n) { return dim3((n + 255) / 256); };

    // ---- SA1 ----
    fps_kernel<<<BATCH, 256, 0, stream>>>(xyz, fidx1, NPTS, 512);
    gather_pts_kernel<<<g1d(8192), 256, 0, stream>>>(xyz, fidx1, l1xyz, NPTS, 512, 8192);
    ball_query_kernel<<<g1d(8192), 256, 0, stream>>>(xyz, l1xyz, nidx1, NPTS, 512, 8192, 0.01f);
    build_g1_kernel<<<g1d(131072), 256, 0, stream>>>(xyz, l1xyz, nidx1, g1A, NPTS, 512, 32, 131072);
    layer(g1A, sa1W0, sa1[0], t1a, nullptr, 131072, 32, 32, 32, 0);
    layer(t1a, sa1W1, sa1[1], g1A, nullptr, 131072, 32, 32, 32, 0);
    layer(g1A, sa1W2, sa1[2], t1c, nullptr, 131072, 32, 64, 64, 0);
    maxpool_kernel<<<g1d(8192 * 64), 256, 0, stream>>>(t1c, l1f, 64, 16, 8192 * 64);

    // ---- SA2 ----
    fps_kernel<<<BATCH, 256, 0, stream>>>(l1xyz, fidx2, 512, 128);
    gather_pts_kernel<<<g1d(2048), 256, 0, stream>>>(l1xyz, fidx2, l2xyz, 512, 128, 2048);
    ball_query_kernel<<<g1d(2048), 256, 0, stream>>>(l1xyz, l2xyz, nidx2, 512, 128, 2048, 0.04f);
    build_g2_kernel<<<g1d(32768), 256, 0, stream>>>(l1xyz, l2xyz, nidx2, l1f, g2A, 512, 128, 64, 96, 32768);
    layer(g2A, sa2W0, sa2[0], t2a, nullptr, 32768, 96, 64, 64, 0);
    layer(t2a, sa2W1, sa2[1], t2b, nullptr, 32768, 64, 128, 128, 0);
    maxpool_kernel<<<g1d(2048 * 128), 256, 0, stream>>>(t2b, l2f, 128, 16, 2048 * 128);

    // ---- SA3 (global) ----
    build_g3_kernel<<<g1d(2048), 256, 0, stream>>>(l2xyz, l2f, g3A, 160, 2048);
    layer(g3A, sa3W0, sa3[0], t3a, nullptr, 2048, 160, 128, 128, 0);
    layer(t3a, sa3W1, sa3[1], t3b, nullptr, 2048, 128, 256, 256, 0);
    maxpool_kernel<<<g1d(16 * 256), 256, 0, stream>>>(t3b, l3f, 256, 128, 16 * 256);

    // ---- FP3 ----
    build_fp3in_kernel<<<g1d(2048), 256, 0, stream>>>(l3f, l2f, fp3in, 2048);
    layer(fp3in, fp3W0, fp3[0], fp3t, nullptr, 2048, 384, 256, 256, 0);
    layer(fp3t, fp3W1, fp3[1], f2buf, nullptr, 2048, 256, 256, 256, 0);

    // ---- FP2 ----
    three_interp_kernel<<<g1d(8192), 256, 0, stream>>>(l1xyz, l2xyz, f2buf, fp2in, 512, 128, 256, 320, 8192);
    append_feats_kernel<<<g1d(8192 * 64), 256, 0, stream>>>(l1f, fp2in, 64, 64, 320, 256, 8192 * 64);
    layer(fp2in, fp2W0, fp2[0], fp2t, nullptr, 8192, 320, 256, 256, 0);
    layer(fp2t, fp2W1, fp2[1], f1buf, nullptr, 8192, 256, 256, 256, 0);

    // ---- FP1 + heads (the big 131072-row stage) ----
    three_interp_kernel<<<g1d(131072), 256, 0, stream>>>(xyz, l1xyz, f1buf, fp1in, NPTS, 512, 256, 256, 131072);
    layer(fp1in, fp1W0, fp1[0], bigA, nullptr, 131072, 256, 128, 128, 0);
    layer(bigA, fp1W1, fp1[1], bigB, nullptr, 131072, 128, 128, 128, 0);
    layer(bigB, fp1W2, fp1[2], bigA, nullptr, 131072, 128, 128, 128, 0);
    layer(bigA, h1W, h1, bigB, nullptr, 131072, 128, 128, 128, 0);
    layer(bigB, h2W, h2, nullptr, (float*)d_out, 131072, 128, 64, 64, 1);
}
